// Steerable_Encoder_28776280883895
// MI455X (gfx1250) — matrix operand order repaired and checked by reading, not compiled, not run on hardware
//
#include <hip/hip_runtime.h>
#include <stdint.h>

// CDNA5 / gfx1250 fused Gaussian-feature-map kernel.
// FM[g, 0..2] = sum_n exp(-|grid_g - X_n|^2 / 2) * [1, Yx_n, Yy_n]
// out[0, c, j, i] = c==0 ? FM0 : FMc / FM0,  g = i*128 + j.
//
// exp(-d2/2) = exp2(tg + q_n + px*Xx + py*Xy) with
//   tg = -L*(gx^2+gy^2), q_n = -L*(Xx^2+Xy^2), px = 2L*gx, py = 2L*gy,
//   L = log2(e)/2.

typedef __attribute__((ext_vector_type(16))) _Float16 v16h;
typedef __attribute__((ext_vector_type(2)))  _Float16 h2;
typedef __attribute__((ext_vector_type(8)))  float    v8f;

#define N_CTX   20000
#define NGRID   16384                 // 128 x 128
#define CH      512                   // context chunk staged in LDS
#define NGROUP  (CH / 32)             // 16 WMMA K-steps per chunk
#define WAVES   4                     // waves per block (wave32)
#define BLOCK   (WAVES * 32)

#define L2E_HALF 0.72134752044448170f // log2(e)/2

__global__ __launch_bounds__(BLOCK) void steerable_encoder_wmma(
    const float* __restrict__ Xp,   // (N_CTX, 2)
    const float* __restrict__ Yp,   // (N_CTX, 2)
    float* __restrict__ out)        // (1, 3, 128, 128)
{
    __shared__ __align__(16) float2 xs[CH];          // raw X chunk      (4 KB)
    __shared__ __align__(16) float  qv[CH];          // -L*|X|^2         (2 KB)
    __shared__ __align__(32) h2 bbuf[NGROUP][32][8]; // packed B tiles  (16 KB)
    __shared__ float osc[WAVES][16][16];             // D writeback      (4 KB)

    const int tid  = threadIdx.x;
    const int lane = tid & 31;
    const int wave = tid >> 5;
    const int half = lane >> 4;       // 0: lanes 0-15, 1: lanes 16-31
    const int m    = lane & 15;       // M row (both halves cover M=0..15)

    // This wave's 16 grid points: g = tile*16 + m
    const int tile = blockIdx.x * WAVES + wave;
    const int g    = tile * 16 + m;
    const int gi   = g >> 7;          // x index
    const int gj   = g & 127;         // y index
    const float gx = -10.0f + (20.0f / 127.0f) * (float)gi;
    const float gy =  10.0f - (20.0f / 127.0f) * (float)gj;

    const float tg = -L2E_HALF * (gx * gx + gy * gy);
    const float px = 2.0f * L2E_HALF * gx;
    const float py = 2.0f * L2E_HALF * gy;

    const float2* __restrict__ X2 = (const float2*)Xp;
    const float2* __restrict__ Y2 = (const float2*)Yp;

    v8f acc = {};   // 16x16 f32 C/D tile (cols 0..2 meaningful)

    const int nchunks = (N_CTX + CH - 1) / CH;   // 40
    for (int chunk = 0; chunk < nchunks; ++chunk) {
        const int base = chunk * CH;

        // ---- stage raw X chunk ----
        if (base + CH <= N_CTX) {
            // Full chunk: ASYNCcnt-tracked DMA, 16B per lane per issue.
            for (int t = tid; t < CH / 2; t += BLOCK) {
                const uint32_t ldsa =
                    (uint32_t)(uintptr_t)(&xs[2 * t]);          // LDS byte addr
                const uint32_t goff =
                    (uint32_t)((base + 2 * t) * sizeof(float2)); // byte offset
                asm volatile(
                    "global_load_async_to_lds_b128 %0, %1, %2"
                    :: "v"(ldsa), "v"(goff), "s"(X2)
                    : "memory");
            }
            asm volatile("s_wait_asynccnt 0x0" ::: "memory");
        } else {
            for (int t = tid; t < CH; t += BLOCK) {
                const int n = base + t;
                xs[t] = (n < N_CTX) ? X2[n] : make_float2(0.0f, 0.0f);
            }
        }

        // ---- stage q_n = -L*|X_n|^2 (zero-pad tail) ----
        for (int t = tid; t < CH; t += BLOCK) {
            const int n = base + t;
            float xx = 0.0f, xy = 0.0f;
            if (n < N_CTX) { const float2 v = X2[n]; xx = v.x; xy = v.y; }
            qv[t] = -L2E_HALF * (xx * xx + xy * xy);
        }

        // ---- stage packed B = [1, Yx, Yy, 0...] in ISA 16-bit B 32x16 layout
        //      lanes 0-15 hold K in {0..7,16..23}, lanes 16-31 K in {8..15,24..31}
        for (int d = tid; d < NGROUP * 32 * 8; d += BLOCK) {
            const int grp = d >> 8;          // / (32*8)
            const int r   = d & 255;
            const int bl  = r >> 3;          // lane slot 0..31
            const int v   = r & 7;           // VGPR slot 0..7
            const int c   = bl & 15;         // N (channel) column
            const int kk0 = 2 * v + ((v >= 4) ? 8 : 0) + 8 * (bl >> 4);
            const int n0  = base + grp * 32 + kk0;
            const int n1  = n0 + 1;
            float w0 = 0.0f, w1 = 0.0f;
            if (n0 < N_CTX)
                w0 = (c == 0) ? 1.0f : (c == 1) ? Y2[n0].x : (c == 2) ? Y2[n0].y : 0.0f;
            if (n1 < N_CTX)
                w1 = (c == 0) ? 1.0f : (c == 1) ? Y2[n1].x : (c == 2) ? Y2[n1].y : 0.0f;
            h2 p;
            p[0] = (_Float16)w0;
            p[1] = (_Float16)w1;
            bbuf[grp][bl][v] = p;
        }
        __syncthreads();

        // ---- inner K loop: one v_wmma_f32_16x16x32_f16 per 32 context pts ----
        #pragma unroll 4
        for (int s = 0; s < NGROUP; ++s) {
            v16h a;
            const int kbase = s * 32;
            #pragma unroll
            for (int v = 0; v < 8; ++v) {
                // ISA 16-bit A 16x32 layout: lanes0-15 K in {0..7,16..23},
                // lanes16-31 K in {8..15,24..31}; VGPR v holds pair (kA, kA+1).
                const int kA = 2 * v + ((v >= 4) ? 8 : 0) + 8 * half;
                const float4 x01 = *(const float4*)&xs[kbase + kA]; // two float2
                const float2 q01 = *(const float2*)&qv[kbase + kA];
                const float t0 = fmaf(px, x01.x, fmaf(py, x01.y, tg + q01.x));
                const float t1 = fmaf(px, x01.z, fmaf(py, x01.w, tg + q01.y));
                const float e0 = __builtin_amdgcn_exp2f(t0);
                const float e1 = __builtin_amdgcn_exp2f(t1);
                a[2 * v]     = (_Float16)e0;
                a[2 * v + 1] = (_Float16)e1;
            }
            const v16h b = *(const v16h*)&bbuf[s][lane][0];
            // 8 args: (neg_a, A, neg_b, B, c_mod, C, reuse_a, reuse_b)
            acc = __builtin_amdgcn_wmma_f32_16x16x32_f16(
                false, a, false, b, (short)0, acc, false, false);
        }
        __syncthreads();
    }

    // ---- spill D tile to LDS per C/D layout, then normalize + store ----
    #pragma unroll
    for (int r = 0; r < 8; ++r) {
        const int M = r + 8 * half;   // lane<16: M=r ; lane>=16: M=r+8
        osc[wave][M][m] = acc[r];     // N = lane&15
    }
    __syncthreads();

    if (lane < 16) {
        const float f0 = osc[wave][m][0];
        const float f1 = osc[wave][m][1];
        const float f2 = osc[wave][m][2];
        const int o = gj * 128 + gi;              // transpose(2,1,0)
        out[o]             = f0;
        out[NGRID + o]     = f1 / f0;
        out[2 * NGRID + o] = f2 / f0;
    }
}

extern "C" void kernel_launch(void* const* d_in, const int* in_sizes, int n_in,
                              void* d_out, int out_size, void* d_ws, size_t ws_size,
                              hipStream_t stream) {
    (void)in_sizes; (void)n_in; (void)d_ws; (void)ws_size; (void)out_size;
    const float* X = (const float*)d_in[0];
    const float* Y = (const float*)d_in[1];
    float* out = (float*)d_out;

    const int ntiles = NGRID / 16;          // 1024 M-tiles
    const int blocks = ntiles / WAVES;      // 256 blocks of 4 waves
    steerable_encoder_wmma<<<blocks, BLOCK, 0, stream>>>(X, Y, out);
}
